// PatchMMConvolution_89404039234220
// MI455X (gfx1250) — matrix operand order repaired and checked by reading, not compiled, not run on hardware
//
#include <hip/hip_runtime.h>

// ---------------------------------------------------------------------------
// PatchMMConvolution on MI455X (gfx1250):
//   out[n,co,oh,ow] = bias[co] + sum_{c,kh,kw} x[n,c,oh+kh-1,ow+kw-1] * w[co,c,kh,kw]
// 36 images, C=64, H=W=128, Cout=128, 3x3, pad 1, f32 in/out.
//
// Implicit GEMM, bf16 data / f32 accumulate via v_wmma_f32_16x16x32_bf16.
// Roofline: 87 GFLOP vs ~454 MB min HBM traffic (~20 us at 23.3 TB/s); bf16
// WMMA is the only matrix path fast enough to reach that floor.
// Block = one full output row (128 ow) x all 128 Cout; 8 waves (4M x 2N),
// each wave 32M x 64N = 144 WMMA, 1.5 loads/WMMA.
// ---------------------------------------------------------------------------

typedef __attribute__((ext_vector_type(16))) __bf16 v16bf;
typedef __attribute__((ext_vector_type(8)))  float  v8f;

#define NIMG   36
#define CIN    64
#define HW     128
#define COUT   128
#define KTOT   576      // 3*3*64
#define KSEG   192      // kh(3)*c(64) per kw
#define XCOLS  130      // 128 + 2 halo columns (always zero for full-row tile)
#define XSTR   200      // LDS row stride in shorts: 400B -> conflict-free A loads
#define OSTR   132      // f32 out-staging row stride: conflict-free b128 spills

__device__ __forceinline__ unsigned short f32_to_bf16_rne(float f) {
    unsigned int u = __float_as_uint(f);
    u = (u + 0x7FFFu + ((u >> 16) & 1u)) >> 16;   // round to nearest even
    return (unsigned short)u;
}

// 32-byte WMMA fragment viewed as two 16B chunks.
typedef union alignas(32) {
    uint4  q[2];
    v16bf  v;
} frag_t;

// ---------------------------------------------------------------------------
// Prep: weight[Cout,C,3,3] f32 -> wsB[co][kw][kh][c] bf16 (147456 B, L2-resident).
// ---------------------------------------------------------------------------
__global__ void prep_weights_kernel(const float* __restrict__ w,
                                    unsigned short* __restrict__ wsB) {
    int idx = blockIdx.x * blockDim.x + threadIdx.x;
    if (idx >= COUT * KTOT) return;
    int co = idx / KTOT;
    int k  = idx - co * KTOT;
    int kw = k / KSEG;
    int r  = k - kw * KSEG;
    int kh = r >> 6;
    int c  = r & 63;
    wsB[idx] = f32_to_bf16_rne(w[((co * CIN + c) * 3 + kh) * 3 + kw]);
}

// ---------------------------------------------------------------------------
// Main kernel: one block per (image, oh). 256 threads = 8 waves.
// ---------------------------------------------------------------------------
__global__ __launch_bounds__(256)
void patch_conv_wmma_kernel(const float* __restrict__ x,
                            const unsigned short* __restrict__ wsB,
                            const float* __restrict__ bias,
                            float* __restrict__ out) {
    // 52000 B LDS: phase 1 = bf16 halo tile [130][200]; phase 2 overlays a
    // 64x132 f32 staging buffer (33792 B) for coalesced output writes.
    __shared__ alignas(16) unsigned char smem[XCOLS * XSTR * 2];
    unsigned short* xlds = (unsigned short*)smem;

    int bid = blockIdx.x;            // 4608 = 36 * 128
    int oh  = bid & 127;
    int n   = bid >> 7;
    int tid = threadIdx.x;

    // Halo columns 0 and 129 are always W-out-of-range for a full-row tile.
    if (tid < 2 * KSEG) {
        int col = (tid < KSEG) ? 0 : (XCOLS - 1);
        int kc  = (tid < KSEG) ? tid : (tid - KSEG);
        xlds[col * XSTR + kc] = 0;
    }

    // Stage interior: xlds[wcol][kh*64+c], 4 channel-consecutive bf16 per pack.
    const float* xbase = x + (size_t)n * (CIN * HW * HW);
    for (int i = tid; i < 128 * 48; i += 256) {
        int kcg  = i >> 7;               // 0..47  (group of 4 kc)
        int wcol = 1 + (i & 127);        // lanes -> consecutive w (coalesced)
        int kc0  = kcg * 4;
        int kh   = kc0 >> 6;
        int c0   = kc0 & 63;
        int h    = oh + kh - 1;
        unsigned short p0 = 0, p1 = 0, p2 = 0, p3 = 0;
        if ((unsigned)h < (unsigned)HW) {
            const float* sp = xbase + ((size_t)c0 * HW + h) * HW + (wcol - 1);
            p0 = f32_to_bf16_rne(sp[0 * HW * HW]);
            p1 = f32_to_bf16_rne(sp[1 * HW * HW]);
            p2 = f32_to_bf16_rne(sp[2 * HW * HW]);
            p3 = f32_to_bf16_rne(sp[3 * HW * HW]);
        }
        uint2 pk;
        pk.x = (unsigned)p0 | ((unsigned)p1 << 16);
        pk.y = (unsigned)p2 | ((unsigned)p3 << 16);
        *(uint2*)(xlds + wcol * XSTR + kc0) = pk;
    }
    __syncthreads();

    int lane  = tid & 31;
    int wave  = tid >> 5;
    int m0    = (wave & 3) * 32;     // 0,32,64,96
    int nhalf = wave >> 2;           // 0,1
    int n0    = nhalf * 64;
    int nm    = lane & 15;           // N col (B/C/D) == M row (A) for this lane
    int khalf = lane >> 4;

    v8f acc[2][4];
    #pragma unroll
    for (int s = 0; s < 2; ++s)
        #pragma unroll
        for (int t = 0; t < 4; ++t) acc[s][t] = (v8f)(0.0f);

    // ---- K loop: kw(3) x six K=32 steps; 8 WMMA per step (B reused 2x). ----
    #pragma unroll
    for (int kw = 0; kw < 3; ++kw) {
        const unsigned short* a0p = xlds + (m0 + nm + kw) * XSTR;
        const unsigned short* a1p = a0p + 16 * XSTR;
        const unsigned short* bb  = wsB + kw * KSEG;
        #pragma unroll
        for (int ks = 0; ks < 6; ++ks) {
            int k0 = ks * 32;
            frag_t a0, a1;   // 16-bit A layout: lane half K {0-7,16-23}/{8-15,24-31}
            a0.q[0] = *(const uint4*)(a0p + k0 + khalf * 8);
            a0.q[1] = *(const uint4*)(a0p + k0 + khalf * 8 + 16);
            a1.q[0] = *(const uint4*)(a1p + k0 + khalf * 8);
            a1.q[1] = *(const uint4*)(a1p + k0 + khalf * 8 + 16);
            #pragma unroll
            for (int t = 0; t < 4; ++t) {
                int co = n0 + t * 16 + nm;
                // B layout: lane half K {0-7,16-23}/{8-15,24-31} (matches A).
                const unsigned short* bp = bb + (size_t)co * KTOT + k0 + khalf * 8;
                frag_t b;
                b.q[0] = *(const uint4*)(bp);
                b.q[1] = *(const uint4*)(bp + 16);
                if (t == 0 && ks < 5) __builtin_prefetch(bp + 32, 0, 0);
                acc[0][t] = __builtin_amdgcn_wmma_f32_16x16x32_bf16(
                    false, a0.v, false, b.v, (short)0, acc[0][t], false, false);
                acc[1][t] = __builtin_amdgcn_wmma_f32_16x16x32_bf16(
                    false, a1.v, false, b.v, (short)0, acc[1][t], false, false);
            }
        }
    }

    // ---- Epilogue: two 64-Cout half-passes through padded f32 LDS tile. ----
    float* outlds = (float*)smem;
    size_t obase = ((size_t)n * COUT) * (HW * HW) + (size_t)oh * HW;
    #pragma unroll
    for (int half = 0; half < 2; ++half) {
        __syncthreads();                 // xlds dead / previous pass streamed
        if (nhalf == half) {
            #pragma unroll
            for (int s = 0; s < 2; ++s) {
                #pragma unroll
                for (int t = 0; t < 4; ++t) {
                    int col = t * 16 + nm;                 // local co 0..63
                    int mb  = m0 + s * 16 + khalf * 8;     // C/D: M = r + 8*khalf
                    float4 lo = make_float4(acc[s][t][0], acc[s][t][1],
                                            acc[s][t][2], acc[s][t][3]);
                    float4 hi = make_float4(acc[s][t][4], acc[s][t][5],
                                            acc[s][t][6], acc[s][t][7]);
                    *(float4*)(outlds + col * OSTR + mb)     = lo;
                    *(float4*)(outlds + col * OSTR + mb + 4) = hi;
                }
            }
        }
        __syncthreads();
        // Stream 64 x 128 f32 to global as 16B stores, 512B-contiguous per co.
        for (int j = tid; j < 64 * (HW / 4); j += 256) {   // 2048
            int col = j >> 5;
            int m4  = j & 31;
            int co  = half * 64 + col;
            float4 v = *(const float4*)(outlds + col * OSTR + m4 * 4);
            float  bv = bias[co];
            v.x += bv; v.y += bv; v.z += bv; v.w += bv;
            *(float4*)(out + obase + (size_t)co * (HW * HW) + m4 * 4) = v;
        }
    }
}

// ---------------------------------------------------------------------------
extern "C" void kernel_launch(void* const* d_in, const int* in_sizes, int n_in,
                              void* d_out, int out_size, void* d_ws, size_t ws_size,
                              hipStream_t stream) {
    const float* x    = (const float*)d_in[0];   // [2,18,64,128,128]
    const float* w    = (const float*)d_in[1];   // [128,64,3,3]
    const float* bias = (const float*)d_in[2];   // [128]
    float*       out  = (float*)d_out;           // [2,18,128,128,128]
    unsigned short* wsB = (unsigned short*)d_ws; // 147456 B bf16 weights

    int wn = COUT * KTOT;                                    // 73728
    prep_weights_kernel<<<(wn + 255) / 256, 256, 0, stream>>>(w, wsB);

    int nblocks = NIMG * HW;                                 // 4608
    patch_conv_wmma_kernel<<<nblocks, 256, 0, stream>>>(x, wsB, bias, out);
}
